// model_71906342470267
// MI455X (gfx1250) — matrix operand order repaired and checked by reading, compiled, not run on hardware
//
#include <hip/hip_runtime.h>
#include <hip/hip_bf16.h>
#include <math.h>

typedef __attribute__((ext_vector_type(16))) _Float16 v16h;
typedef __attribute__((ext_vector_type(8)))  float    v8f;

#define B_    10
#define NG_   2000
#define DIN_  512
#define H_    100
#define NCLS_ 12
#define N_    20000
#define E_    640000
#define HP_   128      // H padded to 4x32 for WMMA K-steps / 8x16 col tiles
#define T_    200      // LSTM sequence length = 2*H

static inline int ceildiv_l(long a, int b){ return (int)((a + b - 1) / b); }

__device__ __forceinline__ float sigf(float x){ return 1.0f/(1.0f + expf(-x)); }

// ---------------- utility fills ----------------
__global__ void k_fill_f32(float* p, float v, long n){
  long i = (long)blockIdx.x*blockDim.x + threadIdx.x; if(i<n) p[i]=v;
}
__global__ void k_fill_i32(int* p, int v, long n){
  long i = (long)blockIdx.x*blockDim.x + threadIdx.x; if(i<n) p[i]=v;
}
__global__ void k_init_edges(const int* __restrict__ ei, int* src, int* dst, float* mask){
  int e = blockIdx.x*blockDim.x + threadIdx.x; if(e>=E_) return;
  src[e]=ei[e]; dst[e]=ei[E_+e]; mask[e]=1.0f;
}
__global__ void k_init_orig(int* o){
  int i = blockIdx.x*blockDim.x + threadIdx.x; if(i<N_) return_void: ;
  if(i<N_) o[i] = i % NG_;
}

// ---------------- WMMA fragment packing ----------------
// A fragment (16x32 f16, MxK): lanes 0-15 row=lane, halves 0..7 -> K off..off+7,
// halves 8..15 -> K off+16..off+23, off = (lane>=16)?8:0.
__global__ void k_pack_a(const float* __restrict__ X, int ldx, int nrows, int ksteps,
                         _Float16* __restrict__ out, long total){
  long t = (long)blockIdx.x*blockDim.x + threadIdx.x; if(t>=total) return;
  int lane = (int)(t & 31); long q = t >> 5;
  int kb = (int)(q % ksteps); long mt = q / ksteps;
  int row = (int)(mt*16 + (lane & 15));
  int off = (lane >> 4) * 8;
  int k0  = kb*32 + off;
  _Float16* o = out + t*16;
  if(row < nrows){
    const float* xr = X + (long)row*ldx + k0;
    #pragma unroll
    for(int j=0;j<8;j++) o[j]   = (_Float16)xr[j];
    #pragma unroll
    for(int j=0;j<8;j++) o[8+j] = (_Float16)xr[16+j];
  } else {
    #pragma unroll
    for(int j=0;j<16;j++) o[j] = (_Float16)0.0f;
  }
}
// B fragment (32x16 f16, KxN): lane col = lane&15, halves j<8 -> K 8*(lane>>4)+j, j>=8 -> K 16+8*(lane>>4)+(j-8)
__global__ void k_pack_b(const float* __restrict__ W, int Krows, int Ncols, int ksteps,
                         _Float16* __restrict__ out, long total){
  long t = (long)blockIdx.x*blockDim.x + threadIdx.x; if(t>=total) return;
  int lane = (int)(t & 31); long q = t >> 5;
  int kb = (int)(q % ksteps); int nt = (int)(q / ksteps);
  int col = nt*16 + (lane & 15);
  _Float16* o = out + t*16;
  #pragma unroll
  for(int j=0;j<16;j++){
    int k = kb*32 + 8*(lane>>4) + (j < 8 ? j : j + 8);
    float v = (k < Krows && col < Ncols) ? W[(long)k*Ncols + col] : 0.0f;
    o[j] = (_Float16)v;
  }
}

// ---------------- WMMA GEMM: C[M x 128] = A[M x 32*KSTEPS] * B ----------------
template<int KSTEPS>
__global__ void k_gemm(const _Float16* __restrict__ Ap, const _Float16* __restrict__ Bp,
                       float* __restrict__ C){
  int lane = threadIdx.x & 31;
  int wv   = threadIdx.x >> 5;            // 8 waves -> 8 column tiles (128 cols)
  const v16h* Ab = (const v16h*)Ap + ((long)blockIdx.x*KSTEPS)*32 + lane;
  const v16h* Bb = (const v16h*)Bp + ((long)wv*KSTEPS)*32 + lane;
  v8f acc = {};
  #pragma unroll
  for(int kb=0; kb<KSTEPS; ++kb){
    if(kb+1 < KSTEPS) __builtin_prefetch((const void*)(Ab + (kb+1)*32), 0, 1);
    v16h a = Ab[kb*32];
    v16h b = Bb[kb*32];
    acc = __builtin_amdgcn_wmma_f32_16x16x32_f16(false, a, false, b, (short)0, acc,
                                                 false, false);
  }
  int row0 = blockIdx.x*16 + 8*(lane>>4);
  int col  = wv*16 + (lane & 15);
  #pragma unroll
  for(int v=0; v<8; ++v) C[(long)(row0+v)*HP_ + col] = acc[v];
}

// ---------------- GCN aggregation ----------------
__global__ void k_deg_accum(const int* __restrict__ dst, const float* __restrict__ ea,
                            const float* __restrict__ mask, float* deg){
  int e = blockIdx.x*blockDim.x + threadIdx.x; if(e>=E_) return;
  float w = ea[e]*mask[e];
  if(w != 0.0f) atomicAdd(&deg[dst[e]], w);
}
__global__ void k_dinv(const float* __restrict__ deg, float* dinv, int n){
  int i = blockIdx.x*blockDim.x + threadIdx.x; if(i>=n) return;
  float d = deg[i];
  dinv[i] = (d > 0.0f) ? rsqrtf(d) : 0.0f;
}
__global__ void k_edge_agg(const int* __restrict__ src, const int* __restrict__ dst,
                           const float* __restrict__ ea, const float* __restrict__ mask,
                           const float* __restrict__ dinv, const float* __restrict__ h,
                           float* __restrict__ agg){
  long t = (long)blockIdx.x*blockDim.x + threadIdx.x;
  if(t >= (long)E_*25) return;
  int e  = (int)(t/25);
  int c4 = (int)(t%25)*4;
  float w = ea[e]*mask[e];
  if(w == 0.0f) return;
  int s = src[e], d = dst[e];
  float nrm = dinv[s]*w*dinv[d];
  if(nrm == 0.0f) return;
  const float* hs = h + (long)s*HP_ + c4;
  float* ad = agg + (long)d*HP_ + c4;
  atomicAdd(ad+0, nrm*hs[0]);
  atomicAdd(ad+1, nrm*hs[1]);
  atomicAdd(ad+2, nrm*hs[2]);
  atomicAdd(ad+3, nrm*hs[3]);
}
__global__ void k_gcn_finish(const float* __restrict__ agg, const float* __restrict__ h,
                             const float* __restrict__ dinv, const float* __restrict__ bias,
                             float* __restrict__ xo, int n){
  long t = (long)blockIdx.x*blockDim.x + threadIdx.x;
  if(t >= (long)n*HP_) return;
  int i = (int)(t/HP_), f = (int)(t%HP_);
  float v = 0.0f;
  if(f < H_){
    float di = dinv[i];
    v = agg[t] + di*di*h[t] + bias[f];     // self-loop contribution + bias
    v = fmaxf(v, 0.0f);                    // relu
  }
  xo[t] = v;
}

// ---------------- SAGPool ----------------
__global__ void k_node_dots(const float* __restrict__ x, const float* __restrict__ wrel,
                            const float* __restrict__ wroot, float* sa, float* sr, int n){
  int i = blockIdx.x*blockDim.x + threadIdx.x; if(i>=n) return;
  const float* xr = x + (long)i*HP_;
  float a=0.0f, r=0.0f;
  for(int f=0; f<H_; ++f){ float v = xr[f]; a += v*wrel[f]; r += v*wroot[f]; }
  sa[i]=a; sr[i]=r;
}
__global__ void k_score_accum(const int* __restrict__ src, const int* __restrict__ dst,
                              const float* __restrict__ mask, const float* __restrict__ sa,
                              float* ssum){
  int e = blockIdx.x*blockDim.x + threadIdx.x; if(e>=E_) return;
  float w = mask[e];
  if(w != 0.0f) atomicAdd(&ssum[dst[e]], w*sa[src[e]]);
}
__global__ void k_score_final(const float* __restrict__ ssum, const float* __restrict__ sr,
                              const float* __restrict__ brel, float* score, int n){
  int i = blockIdx.x*blockDim.x + threadIdx.x; if(i>=n) return;
  score[i] = tanhf(ssum[i] + brel[0] + sr[i]);
}
// exact top-k (sorted descending, stable ties) via rank counting in LDS
__global__ void k_topk(const float* __restrict__ score, int ng, int k, int* perm){
  __shared__ float s[2048];
  int g = blockIdx.x;
  const float* sg = score + (long)g*ng;
  for(int j=threadIdx.x; j<ng; j+=blockDim.x) s[j] = sg[j];
  __syncthreads();
  for(int j=threadIdx.x; j<ng; j+=blockDim.x){
    float v = s[j]; int rank = 0;
    for(int t=0; t<ng; ++t){
      float u = s[t];
      rank += (u > v) || (u == v && t < j);
    }
    if(rank < k) perm[(long)g*k + rank] = g*ng + j;
  }
}
__global__ void k_apply_perm(const float* __restrict__ xin, const float* __restrict__ score,
                             const int* __restrict__ perm, const int* __restrict__ oin,
                             float* __restrict__ xout, int* __restrict__ inv,
                             int* __restrict__ oout, int nnew){
  long t = (long)blockIdx.x*blockDim.x + threadIdx.x;
  if(t >= (long)nnew*HP_) return;
  int p = (int)(t/HP_), f = (int)(t%HP_);
  int io = perm[p];
  xout[t] = (f < H_) ? xin[(long)io*HP_ + f]*score[io] : 0.0f;
  if(f == 0){ inv[io] = p; oout[p] = oin[io]; }
}
__global__ void k_remap(int* src, int* dst, float* mask, const int* __restrict__ inv){
  int e = blockIdx.x*blockDim.x + threadIdx.x; if(e>=E_) return;
  int ns = inv[src[e]], nd = inv[dst[e]];
  if(ns < 0 || nd < 0) mask[e] = 0.0f;
  src[e] = (ns < 0) ? 0 : ns;
  dst[e] = (nd < 0) ? 0 : nd;
}
// global avg+max pool, accumulated into readout
__global__ void k_gpool(const float* __restrict__ x, int k, float* __restrict__ ro){
  int g = blockIdx.x, f = threadIdx.x;
  if(f >= H_) return;
  float sum = 0.0f, mx = -INFINITY;
  const float* base = x + (long)g*k*HP_ + f;
  for(int p=0; p<k; ++p){ float v = base[(long)p*HP_]; sum += v; mx = fmaxf(mx, v); }
  ro[g*T_ + f]      += sum/(float)k;
  ro[g*T_ + H_ + f] += mx;
}

// ---------------- tiny bidirectional 2-layer LSTM (hidden=1) ----------------
__global__ void k_lstm(const float* __restrict__ ro,
    const float* w0fI, const float* w0fH, const float* b0f,
    const float* w0bI, const float* w0bH, const float* b0b,
    const float* w1fI, const float* w1fH, const float* b1f,
    const float* w1bI, const float* w1bH, const float* b1b,
    float* h0f, float* h0b, float* h1f, float* h1b, float* __restrict__ m){
  int b = threadIdx.x;
  if(b >= B_) return;
  float h, c;
  // layer0 forward
  h = 0.0f; c = 0.0f;
  for(int t=0; t<T_; ++t){
    float x = ro[b*T_ + t];
    float gi = w0fI[0]*x + w0fH[0]*h + b0f[0];
    float gf = w0fI[1]*x + w0fH[1]*h + b0f[1];
    float gg = w0fI[2]*x + w0fH[2]*h + b0f[2];
    float go = w0fI[3]*x + w0fH[3]*h + b0f[3];
    c = sigf(gf)*c + sigf(gi)*tanhf(gg);
    h = sigf(go)*tanhf(c);
    h0f[t*B_ + b] = h;
  }
  // layer0 backward
  h = 0.0f; c = 0.0f;
  for(int t=T_-1; t>=0; --t){
    float x = ro[b*T_ + t];
    float gi = w0bI[0]*x + w0bH[0]*h + b0b[0];
    float gf = w0bI[1]*x + w0bH[1]*h + b0b[1];
    float gg = w0bI[2]*x + w0bH[2]*h + b0b[2];
    float go = w0bI[3]*x + w0bH[3]*h + b0b[3];
    c = sigf(gf)*c + sigf(gi)*tanhf(gg);
    h = sigf(go)*tanhf(c);
    h0b[t*B_ + b] = h;
  }
  // layer1 forward (input dim 2)
  h = 0.0f; c = 0.0f;
  for(int t=0; t<T_; ++t){
    float x0 = h0f[t*B_ + b], x1 = h0b[t*B_ + b];
    float gi = w1fI[0]*x0 + w1fI[1]*x1 + w1fH[0]*h + b1f[0];
    float gf = w1fI[2]*x0 + w1fI[3]*x1 + w1fH[1]*h + b1f[1];
    float gg = w1fI[4]*x0 + w1fI[5]*x1 + w1fH[2]*h + b1f[2];
    float go = w1fI[6]*x0 + w1fI[7]*x1 + w1fH[3]*h + b1f[3];
    c = sigf(gf)*c + sigf(gi)*tanhf(gg);
    h = sigf(go)*tanhf(c);
    h1f[t*B_ + b] = h;
  }
  // layer1 backward
  h = 0.0f; c = 0.0f;
  for(int t=T_-1; t>=0; --t){
    float x0 = h0f[t*B_ + b], x1 = h0b[t*B_ + b];
    float gi = w1bI[0]*x0 + w1bI[1]*x1 + w1bH[0]*h + b1b[0];
    float gf = w1bI[2]*x0 + w1bI[3]*x1 + w1bH[1]*h + b1b[1];
    float gg = w1bI[4]*x0 + w1bI[5]*x1 + w1bH[2]*h + b1b[2];
    float go = w1bI[6]*x0 + w1bI[7]*x1 + w1bH[3]*h + b1b[3];
    c = sigf(gf)*c + sigf(gi)*tanhf(gg);
    h = sigf(go)*tanhf(c);
    h1b[t*B_ + b] = h;
  }
  for(int t=0; t<T_; ++t) m[b*T_ + t] = 0.5f*(h1f[t*B_ + b] + h1b[t*B_ + b]);
}

// ---------------- head MLP + log_softmax ----------------
__global__ void k_mlp(const float* __restrict__ m,
                      const float* __restrict__ w1, const float* __restrict__ b1,
                      const float* __restrict__ w2, const float* __restrict__ b2,
                      const float* __restrict__ w3, const float* __restrict__ b3,
                      float* __restrict__ out){
  __shared__ float z1[B_*H_];
  __shared__ float z2[B_*(H_/2)];
  __shared__ float lg[B_*NCLS_];
  for(int t=threadIdx.x; t<B_*H_; t+=blockDim.x){
    int r=t/H_, c=t%H_;
    float s=b1[c];
    for(int f=0; f<2*H_; ++f) s += m[r*(2*H_)+f]*w1[f*H_+c];
    z1[t]=fmaxf(s,0.0f);
  }
  __syncthreads();
  for(int t=threadIdx.x; t<B_*(H_/2); t+=blockDim.x){
    int r=t/(H_/2), c=t%(H_/2);
    float s=b2[c];
    for(int f=0; f<H_; ++f) s += z1[r*H_+f]*w2[f*(H_/2)+c];
    z2[t]=fmaxf(s,0.0f);
  }
  __syncthreads();
  for(int t=threadIdx.x; t<B_*NCLS_; t+=blockDim.x){
    int r=t/NCLS_, c=t%NCLS_;
    float s=b3[c];
    for(int f=0; f<H_/2; ++f) s += z2[r*(H_/2)+f]*w3[f*NCLS_+c];
    lg[t]=s;
  }
  __syncthreads();
  if(threadIdx.x < B_){
    int r = threadIdx.x;
    float mx = -INFINITY;
    for(int c=0;c<NCLS_;c++) mx = fmaxf(mx, lg[r*NCLS_+c]);
    float se = 0.0f;
    for(int c=0;c<NCLS_;c++) se += expf(lg[r*NCLS_+c]-mx);
    float lse = mx + logf(se);
    for(int c=0;c<NCLS_;c++) out[r*NCLS_+c] = lg[r*NCLS_+c]-lse;
  }
}
__global__ void k_write_idx(const int* __restrict__ o, int* __restrict__ out, int n){
  int i = blockIdx.x*blockDim.x + threadIdx.x; if(i<n) out[i]=o[i];
}

// ---------------- host orchestration ----------------
extern "C" void kernel_launch(void* const* d_in, const int* in_sizes, int n_in,
                              void* d_out, int out_size, void* d_ws, size_t ws_size,
                              hipStream_t stream){
  (void)in_sizes; (void)n_in; (void)out_size; (void)ws_size;
  const float* x   = (const float*)d_in[0];
  const float* ea  = (const float*)d_in[1];
  const float* Wc[3]   = {(const float*)d_in[2],(const float*)d_in[4],(const float*)d_in[6]};
  const float* Bc[3]   = {(const float*)d_in[3],(const float*)d_in[5],(const float*)d_in[7]};
  const float* Wrel[3] = {(const float*)d_in[8],(const float*)d_in[11],(const float*)d_in[14]};
  const float* Brel[3] = {(const float*)d_in[9],(const float*)d_in[12],(const float*)d_in[15]};
  const float* Wroot[3]= {(const float*)d_in[10],(const float*)d_in[13],(const float*)d_in[16]};
  const float* L[12];
  for(int i=0;i<12;i++) L[i]=(const float*)d_in[17+i];
  const float* wm1=(const float*)d_in[29]; const float* bm1=(const float*)d_in[30];
  const float* wm2=(const float*)d_in[31]; const float* bm2=(const float*)d_in[32];
  const float* wm3=(const float*)d_in[33]; const float* bm3=(const float*)d_in[34];
  const int* eidx = (const int*)d_in[35];

  char* wp = (char*)d_ws;
  auto alloc = [&](size_t bytes)->void*{
    void* p = (void*)wp; wp += (bytes + 255) & ~(size_t)255; return p;
  };
  _Float16* apack = (_Float16*)alloc((size_t)1250*16*32*16*2);
  _Float16* bpack = (_Float16*)alloc((size_t)8*16*32*16*2);
  float* hbuf = (float*)alloc((size_t)N_*HP_*4);
  float* aggb = (float*)alloc((size_t)N_*HP_*4);
  float* xA   = (float*)alloc((size_t)N_*HP_*4);
  float* xB   = (float*)alloc((size_t)N_*HP_*4);
  float* deg  = (float*)alloc((size_t)N_*4);
  float* dinv = (float*)alloc((size_t)N_*4);
  float* sa   = (float*)alloc((size_t)N_*4);
  float* sr   = (float*)alloc((size_t)N_*4);
  float* ssum = (float*)alloc((size_t)N_*4);
  float* score= (float*)alloc((size_t)N_*4);
  int*   perm = (int*)alloc((size_t)10000*4);
  int*   inv  = (int*)alloc((size_t)N_*4);
  int*   origA= (int*)alloc((size_t)N_*4);
  int*   origB= (int*)alloc((size_t)N_*4);
  int*   csrc = (int*)alloc((size_t)E_*4);
  int*   cdst = (int*)alloc((size_t)E_*4);
  float* cmask= (float*)alloc((size_t)E_*4);
  float* ro   = (float*)alloc((size_t)B_*T_*4);
  float* h0f  = (float*)alloc((size_t)T_*B_*4);
  float* h0b  = (float*)alloc((size_t)T_*B_*4);
  float* h1f  = (float*)alloc((size_t)T_*B_*4);
  float* h1b  = (float*)alloc((size_t)T_*B_*4);
  float* mbuf = (float*)alloc((size_t)B_*T_*4);

  const int TB = 256;
  k_init_edges<<<ceildiv_l(E_,TB),TB,0,stream>>>(eidx,csrc,cdst,cmask);
  k_init_orig<<<ceildiv_l(N_,TB),TB,0,stream>>>(origA);
  k_fill_f32<<<ceildiv_l(B_*T_,TB),TB,0,stream>>>(ro,0.0f,(long)B_*T_);

  int* ocur = origA; int* onext = origB;
  const float* Xsrc = x; int ldx = DIN_;
  const int nArr[3]  = {N_, 10000, 5000};
  const int ngArr[3] = {2000, 1000, 500};
  const int ksArr[3] = {16, 4, 4};
  const int mtArr[3] = {1250, 625, 313};
  const int WrowsArr[3] = {512, 100, 100};

  for(int l=0; l<3; ++l){
    int n = nArr[l], ng = ngArr[l], k = ng/2, nk = B_*k;
    int ks = ksArr[l], mt = mtArr[l];
    // --- GCN: h = X @ W (WMMA f16->f32) ---
    long tA = (long)mt*ks*32;
    k_pack_a<<<ceildiv_l(tA,TB),TB,0,stream>>>(Xsrc, ldx, n, ks, apack, tA);
    long tBn = (long)8*ks*32;
    k_pack_b<<<ceildiv_l(tBn,TB),TB,0,stream>>>(Wc[l], WrowsArr[l], H_, ks, bpack, tBn);
    if(ks == 16) k_gemm<16><<<mt,256,0,stream>>>(apack,bpack,hbuf);
    else         k_gemm<4 ><<<mt,256,0,stream>>>(apack,bpack,hbuf);
    // --- symmetric-norm aggregation ---
    k_fill_f32<<<ceildiv_l(n,TB),TB,0,stream>>>(deg,1.0f,(long)n);          // self loops
    k_fill_f32<<<ceildiv_l((long)n*HP_,TB),TB,0,stream>>>(aggb,0.0f,(long)n*HP_);
    k_deg_accum<<<ceildiv_l(E_,TB),TB,0,stream>>>(cdst,ea,cmask,deg);
    k_dinv<<<ceildiv_l(n,TB),TB,0,stream>>>(deg,dinv,n);
    k_edge_agg<<<ceildiv_l((long)E_*25,TB),TB,0,stream>>>(csrc,cdst,ea,cmask,dinv,hbuf,aggb);
    k_gcn_finish<<<ceildiv_l((long)n*HP_,TB),TB,0,stream>>>(aggb,hbuf,dinv,Bc[l],xA,n);
    // --- SAGPool: linearized scoring, top-k, filter ---
    k_node_dots<<<ceildiv_l(n,TB),TB,0,stream>>>(xA,Wrel[l],Wroot[l],sa,sr,n);
    k_fill_f32<<<ceildiv_l(n,TB),TB,0,stream>>>(ssum,0.0f,(long)n);
    k_score_accum<<<ceildiv_l(E_,TB),TB,0,stream>>>(csrc,cdst,cmask,sa,ssum);
    k_score_final<<<ceildiv_l(n,TB),TB,0,stream>>>(ssum,sr,Brel[l],score,n);
    k_fill_i32<<<ceildiv_l(n,TB),TB,0,stream>>>(inv,-1,(long)n);
    k_topk<<<B_,256,0,stream>>>(score,ng,k,perm);
    k_apply_perm<<<ceildiv_l((long)nk*HP_,TB),TB,0,stream>>>(xA,score,perm,ocur,xB,inv,onext,nk);
    k_remap<<<ceildiv_l(E_,TB),TB,0,stream>>>(csrc,cdst,cmask,inv);
    k_gpool<<<B_,128,0,stream>>>(xB,k,ro);
    { int* t = ocur; ocur = onext; onext = t; }
    Xsrc = xB; ldx = HP_;
  }

  k_lstm<<<1,32,0,stream>>>(ro, L[0],L[1],L[2], L[3],L[4],L[5],
                            L[6],L[7],L[8], L[9],L[10],L[11],
                            h0f,h0b,h1f,h1b, mbuf);
  k_mlp<<<1,256,0,stream>>>(mbuf, wm1,bm1, wm2,bm2, wm3,bm3, (float*)d_out);
  k_write_idx<<<ceildiv_l(2500,TB),TB,0,stream>>>(ocur, ((int*)d_out)+B_*NCLS_, 2500);
}
